// NnBoard768_6828998001412
// MI455X (gfx1250) — matrix operand order repaired and checked by reading, not compiled, not run on hardware
//
#include <hip/hip_runtime.h>
#include <stdint.h>

// ---------------------------------------------------------------------------
// NNUE forward on MI455X (gfx1250): sparse features -> dense bf16 one-hot A,
// feature transform as bf16 WMMA GEMM [2B,768]x[768,512], fused clipped-ReLU +
// output-layer partial dot in the GEMM epilogue.
// Round 2: wave tile enlarged to M=64 x N=64 (16 wmma : 16 b128-loads per
// K-step, ~0.8 GB total L2 traffic vs 1.15 GB at M=32), WGP-scope prefetch.
// ---------------------------------------------------------------------------

typedef __attribute__((ext_vector_type(16))) __bf16 v16bf;
typedef __attribute__((ext_vector_type(8)))  float v8f;

#define BATCH   16384
#define NNZ_PER 32
#define FDIM    768
#define HDIM    512
#define ROWS    (2 * BATCH)          // stm rows [0,B), nstm rows [B,2B)
#define KSTEPS  (FDIM / 32)          // 24
#define MTILES  (ROWS / 16)          // 2048
#define NTILES  (HDIM / 16)          // 32
#define FRAG_BYTES 1024              // 32 lanes x 8 u32 per (tile, kstep)
#define TILE_BYTES (KSTEPS * FRAG_BYTES)           // 24576
#define A_BYTES ((size_t)MTILES * TILE_BYTES)      // 50,331,648
#define W_BYTES ((size_t)NTILES * TILE_BYTES)      // 786,432
#define P_BYTES ((size_t)ROWS * 8 * sizeof(float)) // 1,048,576

__device__ __forceinline__ float bf16_to_f(unsigned short u) {
    union { unsigned u; float f; } x; x.u = ((unsigned)u) << 16; return x.f;
}
__device__ __forceinline__ unsigned short f_to_bf16(float f) {
    union { float f; unsigned u; } x; x.f = f;
    unsigned u = x.u + (0x7FFFu + ((x.u >> 16) & 1u));   // round-to-nearest-even
    return (unsigned short)(u >> 16);
}

// Byte offset of A[row][feat] inside the staged A-fragment buffer.
// 16-bit A 16x32 layout (ISA 7.12.2): lane = M + 16*((K%16)>=8),
// vgpr = (K%8)/2 + 4*(K>=16), half = K&1.
__device__ __forceinline__ size_t a_elem_off(int row, int f) {
    int mtile = row >> 4, M = row & 15;
    int kstep = f >> 5,   K = f & 31;
    int lane = M + (((K & 15) >= 8) ? 16 : 0);
    int vg   = ((K & 7) >> 1) + ((K >= 16) ? 4 : 0);
    int half = K & 1;
    return (size_t)mtile * TILE_BYTES + (size_t)kstep * FRAG_BYTES
         + (size_t)(lane * 32 + vg * 4 + half * 2);
}

// ---------------------------------------------------------------------------
__global__ void k_zero(uint4* p, int n4) {
    for (int i = blockIdx.x * blockDim.x + threadIdx.x; i < n4;
         i += gridDim.x * blockDim.x)
        p[i] = make_uint4(0u, 0u, 0u, 0u);
}

// Stage W_ft (f32 [H,F]) -> bf16 B-fragment layout [ntile][kstep][lane][8 u32].
// B 32x16 layout (matches A): lanes 0-15 hold K=0..7,16..23 (2 per VGPR), lanes 16-31 K=8..15,24..31.
__global__ void k_wstage(const float* __restrict__ Wft, unsigned* __restrict__ wst) {
    int s = blockIdx.x * blockDim.x + threadIdx.x;      // one u32 slot each
    if (s >= NTILES * KSTEPS * 32 * 8) return;
    int ntile = s / (KSTEPS * 256);
    int r     = s % (KSTEPS * 256);
    int kstep = r / 256;
    int r2    = r % 256;
    int lane  = r2 >> 3;
    int vg    = r2 & 7;
    int k0 = kstep * 32 + (lane >> 4) * 8 + (vg & 3) * 2 + (vg >> 2) * 16;    // K of low half
    int n  = ntile * 16 + (lane & 15);
    unsigned short lo = f_to_bf16(Wft[n * FDIM + k0]);
    unsigned short hi = f_to_bf16(Wft[n * FDIM + k0 + 1]);
    wst[s] = (unsigned)lo | ((unsigned)hi << 16);
}

// Build dense bf16 one-hot A directly in fragment layout. One thread per row;
// batch_row = repeat(arange(B), 32) so each thread owns its row's 32 nnz
// exclusively -> duplicate features accumulate serially, no atomics needed.
__global__ void k_scatter(const long long* __restrict__ stm_idx,
                          const long long* __restrict__ nstm_idx,
                          const float* __restrict__ stm_val,
                          const float* __restrict__ nstm_val,
                          unsigned char* __restrict__ abase) {
    int t = blockIdx.x * blockDim.x + threadIdx.x;
    if (t >= ROWS) return;
    bool nstm = (t >= BATCH);
    int b = nstm ? t - BATCH : t;
    const long long* idx = nstm ? nstm_idx : stm_idx;
    const float*     val = nstm ? nstm_val : stm_val;
    const int NNZ = BATCH * NNZ_PER;
    int j0 = b * NNZ_PER;
    for (int k = 0; k < NNZ_PER; ++k) {
        int j   = j0 + k;
        int row = (int)idx[j] + (nstm ? BATCH : 0);
        int f   = (int)idx[NNZ + j];
        unsigned short* p = (unsigned short*)(abase + a_elem_off(row, f));
        *p = f_to_bf16(bf16_to_f(*p) + val[j]);
    }
}

// GEMM + fused epilogue. Wave tile: M=64 (4 A-frags) x N=64 (4 B-frags).
// 16 wmma : 16 b128-loads per K-step. 8 waves/block, 512 blocks ->
// 4096 wave-tiles = (32768/64)*(512/64).
__global__ __launch_bounds__(256, 1) void
k_gemm(const unsigned char* __restrict__ abase,
       const unsigned char* __restrict__ wbase,
       const float* __restrict__ bft,
       const float* __restrict__ wout,
       float* __restrict__ partial2) {
    int lane = threadIdx.x & 31;
    int wid  = blockIdx.x * 8 + (threadIdx.x >> 5);
    int mt   = wid >> 3;            // [0,512)
    int nb   = wid & 7;             // [0,8)
    int rbase = mt * 64;
    int n0    = nb * 64;

    const unsigned char* aptr = abase + (size_t)(mt * 4) * TILE_BYTES + lane * 32;
    const unsigned char* wptr = wbase + (size_t)(nb * 4) * TILE_BYTES + lane * 32;

    v8f acc[4][4] = {};

    for (int ks = 0; ks < KSTEPS; ++ks) {
        if (ks + 1 < KSTEPS) {   // WGP-scope prefetch of next K-slab
            __builtin_prefetch(aptr + (ks + 1) * FRAG_BYTES, 0, 3);
            __builtin_prefetch(wptr + (ks + 1) * FRAG_BYTES, 0, 3);
        }
        v16bf a[4], bb[4];
#pragma unroll
        for (int i = 0; i < 4; ++i)
            a[i] = *(const v16bf*)(aptr + (size_t)i * TILE_BYTES + ks * FRAG_BYTES);
#pragma unroll
        for (int j = 0; j < 4; ++j)
            bb[j] = *(const v16bf*)(wptr + (size_t)j * TILE_BYTES + ks * FRAG_BYTES);
#pragma unroll
        for (int i = 0; i < 4; ++i)
#pragma unroll
            for (int j = 0; j < 4; ++j)
                acc[i][j] = __builtin_amdgcn_wmma_f32_16x16x32_bf16(
                    false, a[i], false, bb[j], (short)0, acc[i][j], false, false);
    }

    // Epilogue: C layout (ISA 7.12.2): VGPR g, lanes 0-15 -> M=g, N=lane;
    // lanes 16-31 -> M=g+8, N=lane-16.
    int Ncol = lane & 15;
    int off  = (rbase < BATCH) ? 0 : HDIM;   // W_out half per perspective
    float bfv[4], wov[4];
#pragma unroll
    for (int j = 0; j < 4; ++j) {
        int h = n0 + j * 16 + Ncol;
        bfv[j] = bft[h];
        wov[j] = wout[off + h];
    }
#pragma unroll
    for (int i = 0; i < 4; ++i) {
#pragma unroll
        for (int g = 0; g < 8; ++g) {
            float v = 0.f;
#pragma unroll
            for (int j = 0; j < 4; ++j) {
                float x = acc[i][j][g] + bfv[j];
                x = fminf(fmaxf(x, 0.f), 1.f);      // clipped ReLU
                v += x * wov[j];
            }
            // reduce over N within each 16-lane half (masks < 16 stay in-group)
            v += __shfl_xor(v, 1, 32);
            v += __shfl_xor(v, 2, 32);
            v += __shfl_xor(v, 4, 32);
            v += __shfl_xor(v, 8, 32);
            if ((lane & 15) == 0) {
                int rowM = rbase + i * 16 + g + ((lane >> 4) << 3);
                partial2[rowM * 8 + nb] = v;
            }
        }
    }
}

__global__ void k_finish(const float* __restrict__ partial2,
                         const float* __restrict__ bout,
                         float* __restrict__ out) {
    int b = blockIdx.x * blockDim.x + threadIdx.x;
    if (b >= BATCH) return;
    float s = bout[0];
#pragma unroll
    for (int c = 0; c < 8; ++c)
        s += partial2[b * 8 + c] + partial2[(BATCH + b) * 8 + c];
    out[b] = 1.f / (1.f + expf(-s));
}

// ---------------------------------------------------------------------------
extern "C" void kernel_launch(void* const* d_in, const int* in_sizes, int n_in,
                              void* d_out, int out_size, void* d_ws, size_t ws_size,
                              hipStream_t stream) {
    const long long* stm_idx  = (const long long*)d_in[0];
    const long long* nstm_idx = (const long long*)d_in[1];
    const float*     stm_val  = (const float*)d_in[2];
    const float*     nstm_val = (const float*)d_in[3];
    // d_in[4] = batch_size (compile-time constant BATCH)
    const float* W_ft  = (const float*)d_in[5];
    const float* b_ft  = (const float*)d_in[6];
    const float* W_out = (const float*)d_in[7];
    const float* b_out = (const float*)d_in[8];
    float* out = (float*)d_out;

    unsigned char* ws = (unsigned char*)d_ws;
    unsigned char* Abase = ws;                       // 50.0 MB, bf16 A fragments
    unsigned char* Wbase = ws + A_BYTES;             // 0.75 MB, bf16 B fragments
    float* partial2 = (float*)(ws + A_BYTES + W_BYTES); // 1 MB row/colblock sums
    // total workspace use: ~52.2 MB

    // 1) zero staged A
    int n4 = (int)(A_BYTES / 16);
    k_zero<<<1024, 256, 0, stream>>>((uint4*)Abase, n4);
    // 2) stage W_ft into WMMA B-fragment layout (bf16)
    k_wstage<<<(NTILES * KSTEPS * 256 + 255) / 256, 256, 0, stream>>>(W_ft, (unsigned*)Wbase);
    // 3) scatter sparse features into A fragments (no atomics; row-exclusive)
    k_scatter<<<(ROWS + 255) / 256, 256, 0, stream>>>(stm_idx, nstm_idx,
                                                      stm_val, nstm_val, Abase);
    // 4) bf16 WMMA GEMM + fused clipped-ReLU + output-layer partial dot
    k_gemm<<<512, 256, 0, stream>>>(Abase, Wbase, b_ft, W_out, partial2);
    // 5) per-position reduction + bias + sigmoid
    k_finish<<<(BATCH + 255) / 256, 256, 0, stream>>>(partial2, b_out, out);
}
